// DeflateVertexToHex_77618648973580
// MI455X (gfx1250) — compile-verified, matrix operand order checked by reading, not run on hardware
//
#include <hip/hip_runtime.h>

// ---------------------------------------------------------------------------
// MI455X (gfx1250, wave32) fused masked-gather mean-pool + bf16 WMMA GEMM.
//   out[b,t,h] = (sum_{j: idx[t,j]>=0} vf[b, idx[t,j], :]) / max(cnt,1) @ W + b
// GEMM view: M = B*T = 40000, K = D = 512, N = H = 512.
// Kernel 1: W (f32, [k][n]) -> Wt (bf16, [n][k]) in d_ws  (one-time, 512 KB)
// Kernel 2: per 64x256 tile: pool rows into LDS (bf16), async-copy Wt tile
//           into LDS via GLOBAL_LOAD_ASYNC_TO_LDS_B128, 8 WMMAs/wave/K-step.
// ---------------------------------------------------------------------------

#define THREADS 256

typedef __attribute__((ext_vector_type(16))) __bf16      v16bf;
typedef __attribute__((ext_vector_type(8)))  __bf16      bf8v;
typedef __attribute__((ext_vector_type(8)))  float       v8f;
typedef __attribute__((ext_vector_type(4)))  float       f4;
typedef __attribute__((ext_vector_type(4)))  unsigned int u4;

constexpr int BB = 4;        // batch
constexpr int NV = 20000;    // vertices
constexpr int DK = 512;      // feature dim (K)
constexpr int TT = 10000;    // hexes
constexpr int HH = 512;      // out dim (N)
constexpr int M  = BB * TT;  // 40000 rows

constexpr int MT = 64;       // M tile
constexpr int NT = 256;      // N tile
constexpr int KT = 32;       // K step (one bf16 WMMA depth)
constexpr int LDA = 40;      // padded LDS stride (80B: 16B-aligned, bank-step 20)

// gfx1250 async global->LDS copy, 16B/lane, tracked by ASYNCcnt.
// ISA: imm offset applies to both LDS and global address -> one base, imm offsets.
__device__ __forceinline__ void async_b128_to_lds(unsigned lds_off, const void* gptr, int ioff) {
    asm volatile("global_load_async_to_lds_b128 %0, %1, off offset:%2"
                 :
                 : "v"(lds_off), "v"(gptr), "n"(ioff)
                 : "memory");
}
__device__ __forceinline__ void wait_asynccnt0() {
    asm volatile("s_wait_asynccnt 0x0" ::: "memory");
}

// ---------------------------------------------------------------------------
// Kernel 1: Wt[n][k] (bf16) = W[k][n] (f32), via 64x64 LDS tile transpose.
// ---------------------------------------------------------------------------
__global__ __launch_bounds__(256)
void w_transpose_bf16(const float* __restrict__ W, __bf16* __restrict__ Wt)
{
    __shared__ __bf16 tile[64][72];           // [n][k], 144B rows (16B aligned)
    const int tid = threadIdx.x;
    const int k0 = blockIdx.x * 64;
    const int n0 = blockIdx.y * 64;

    // Phase 1: coalesced f32 reads of W rows, bf16-convert, scatter into [n][k]
    {
        const int k = tid >> 2;               // 0..63
        const int g = tid & 3;                // 16 n's each
        const float* src = W + (size_t)(k0 + k) * HH + n0 + g * 16;
        f4 v[4];
        #pragma unroll
        for (int q = 0; q < 4; ++q)
            v[q] = *reinterpret_cast<const f4*>(src + q * 4);
        #pragma unroll
        for (int i = 0; i < 16; ++i)
            tile[g * 16 + i][k] = (__bf16)v[i >> 2][i & 3];
    }
    __syncthreads();

    // Phase 2: vectorized 32B writes of Wt rows (contiguous in k)
    {
        const int n = tid >> 2;
        const int g = tid & 3;
        u4 a = *reinterpret_cast<const u4*>(&tile[n][g * 16]);
        u4 b = *reinterpret_cast<const u4*>(&tile[n][g * 16 + 8]);
        __bf16* dst = Wt + (size_t)(n0 + n) * DK + k0 + g * 16;
        *reinterpret_cast<u4*>(dst)     = a;
        *reinterpret_cast<u4*>(dst + 8) = b;
    }
}

// ---------------------------------------------------------------------------
// Kernel 2: fused pool + GEMM.
// ---------------------------------------------------------------------------
__global__ __launch_bounds__(THREADS)
void hexpool_wmma_gemm(const float* __restrict__ vf,
                       const int*   __restrict__ hex,
                       const __bf16* __restrict__ Wt,
                       const float* __restrict__ bias,
                       float*       __restrict__ out)
{
    __shared__ __bf16 As[MT][LDA];   // pooled rows  [m][k]
    __shared__ __bf16 Bs[NT][LDA];   // Wt tile      [n][k]
    __shared__ int    idxs[MT][6];
    __shared__ float  invc[MT];

    const int tid = threadIdx.x;
    const int r0  = blockIdx.x * MT;     // row base (b*T + t)
    const int n0  = blockIdx.y * NT;     // col base (h)

    // --- Prologue: cache hex indices + inverse counts for this row band ---
    for (int i = tid; i < MT * 6; i += THREADS) {
        int row = i / 6, j = i % 6;
        int t = (r0 + row) % TT;
        idxs[row][j] = hex[t * 6 + j];
    }
    __syncthreads();
    if (tid < MT) {
        int c = 0;
        #pragma unroll
        for (int j = 0; j < 6; ++j) c += (idxs[tid][j] >= 0) ? 1 : 0;
        invc[tid] = 1.0f / (float)(c < 1 ? 1 : c);
    }
    __syncthreads();

    // --- Stage-A mapping: thread -> (row, 8-float chunk of K-step) ---
    const int arow   = tid >> 2;                 // 0..63
    const int aq     = tid & 3;                  // 8 floats each
    const int abatch = (r0 + arow) / TT;
    const float* vfb = vf + (size_t)abatch * NV * DK;
    const float  ainv = invc[arow];

    // Branchless gather setup (loop-invariant): clamp index to 0 so the load
    // is always safe, and fold validity * (1/count) into a per-neighbor scale.
    // Invalid neighbors (idx < 0, ~0.005% of entries) contribute exactly 0.
    const float* gp[6];
    float sc[6];
    #pragma unroll
    for (int j = 0; j < 6; ++j) {
        int id = idxs[arow][j];
        sc[j] = (id >= 0) ? ainv : 0.0f;
        int safe = id < 0 ? 0 : id;
        gp[j] = vfb + (size_t)safe * DK + aq * 8;
    }

    // --- Compute mapping: 8 waves = 4 (M) x 2 (N) ---
    const int wave = tid >> 5, lane = tid & 31;
    const int wm   = wave & 3;                   // M subtile: rows wm*16
    const int wn   = wave >> 2;                  // N half:   cols wn*128
    const int half = lane >> 4;                  // K-half selector (ISA layout)
    const int mlan = lane & 15;                  // M (A) / N (B,C/D) lane index

    // Stage-B copy bases: thread tid owns LDS row Bs[tid] (64B per K-step).
    const __bf16* wrow = Wt + (size_t)(n0 + tid) * DK;
    const unsigned bs_off = (unsigned)(uintptr_t)(&Bs[tid][0]);

    v8f acc[8] = {};                             // 8 x 16x16 f32 accumulators

    for (int ko = 0; ko < DK; ko += KT) {
        // ---- Stage B: Wt[n0+tid][ko..ko+32) -> Bs[tid][0..32): 4 async b128,
        //      single base + immediate offsets (added to BOTH LDS and global).
        {
            const void* g = wrow + ko;
            async_b128_to_lds(bs_off, g, 0);
            async_b128_to_lds(bs_off, g, 16);
            async_b128_to_lds(bs_off, g, 32);
            async_b128_to_lds(bs_off, g, 48);
        }

        // ---- Stage A: branchless gather + scaled accumulate -> bf16 As ----
        f4 s0 = {0.f, 0.f, 0.f, 0.f};
        f4 s1 = {0.f, 0.f, 0.f, 0.f};
        #pragma unroll
        for (int j = 0; j < 6; ++j) {
            const float* p = gp[j] + ko;
            f4 x0 = *reinterpret_cast<const f4*>(p);
            f4 x1 = *reinterpret_cast<const f4*>(p + 4);
            s0 += x0 * sc[j];
            s1 += x1 * sc[j];
        }
        bf8v pk;
        pk[0] = (__bf16)s0.x; pk[1] = (__bf16)s0.y;
        pk[2] = (__bf16)s0.z; pk[3] = (__bf16)s0.w;
        pk[4] = (__bf16)s1.x; pk[5] = (__bf16)s1.y;
        pk[6] = (__bf16)s1.z; pk[7] = (__bf16)s1.w;
        *reinterpret_cast<bf8v*>(&As[arow][aq * 8]) = pk;

        wait_asynccnt0();                        // our async LDS writes landed
        __syncthreads();                         // + everyone's ds stores

        // ---- Compute: 8 bf16 WMMAs per wave per K-step ----
        union AF { v16bf v; u4 u[2]; };
        AF af;
        af.u[0] = *reinterpret_cast<const u4*>(&As[wm * 16 + mlan][half * 8]);
        af.u[1] = *reinterpret_cast<const u4*>(&As[wm * 16 + mlan][16 + half * 8]);
        #pragma unroll
        for (int j = 0; j < 8; ++j) {
            AF bfm;
            const int n = wn * 128 + j * 16 + mlan;
            bfm.u[0] = *reinterpret_cast<const u4*>(&Bs[n][half * 8]);
            bfm.u[1] = *reinterpret_cast<const u4*>(&Bs[n][16 + half * 8]);
            acc[j] = __builtin_amdgcn_wmma_f32_16x16x32_bf16(
                         false, af.v, false, bfm.v,
                         (short)0, acc[j], false, false);
        }
        __syncthreads();
    }

    // ---- Epilogue: C/D layout -> global, + bias ----
    // lane in [0,16): M = v, N = lane ; lane in [16,32): M = 8+v, N = lane-16
    const int orow = r0 + wm * 16 + half * 8;
    #pragma unroll
    for (int j = 0; j < 8; ++j) {
        const int col = n0 + wn * 128 + j * 16 + mlan;
        const float bv = bias[col];
        #pragma unroll
        for (int v = 0; v < 8; ++v) {
            out[(size_t)(orow + v) * HH + col] = acc[j][v] + bv;
        }
    }
}

extern "C" void kernel_launch(void* const* d_in, const int* in_sizes, int n_in,
                              void* d_out, int out_size, void* d_ws, size_t ws_size,
                              hipStream_t stream) {
    const float* vf   = (const float*)d_in[0];   // (B, N, D) f32
    const int*   hex  = (const int*)d_in[1];     // (T, 6) int
    const float* W    = (const float*)d_in[2];   // (D, H) f32
    const float* bias = (const float*)d_in[3];   // (H,) f32
    float* out = (float*)d_out;                  // (B, T, H) f32

    __bf16* Wt = (__bf16*)d_ws;                  // 512*512*2 = 512 KB scratch

    dim3 pgrid(DK / 64, HH / 64);                // 8 x 8
    w_transpose_bf16<<<pgrid, 256, 0, stream>>>(W, Wt);

    dim3 grid(M / MT, HH / NT);                  // 625 x 2, exact cover
    hexpool_wmma_gemm<<<grid, THREADS, 0, stream>>>(vf, hex, Wt, bias, out);
}
